// GraphSageWithSampling_18339510354450
// MI455X (gfx1250) — matrix operand order repaired and checked by reading, not compiled, not run on hardware
//
#include <hip/hip_runtime.h>
#include <hip/hip_bf16.h>

typedef __attribute__((ext_vector_type(16))) __bf16 v16bf;
typedef __attribute__((ext_vector_type(8)))  float  v8f;

union BF16x16 { unsigned int u[8]; uint4 q[2]; v16bf v; };

// pack two fp32 into packed bf16 (truncate) with a single v_perm_b32
__device__ __forceinline__ unsigned int pack2bf(float a, float b) {
  unsigned int ua = __builtin_bit_cast(unsigned int, a);
  unsigned int ub = __builtin_bit_cast(unsigned int, b);
  // result bytes [1:0] = a[3:2], [3:2] = b[3:2]
  return __builtin_amdgcn_perm(ub, ua, 0x07060302u);
}

// fp32 -> bf16 pre-conversion for weight matrices (n even)
__global__ __launch_bounds__(256) void cvt_bf16_kernel(
    const float* __restrict__ in, unsigned int* __restrict__ out, int n) {
  const int i = (blockIdx.x * 256 + threadIdx.x) * 2;
  if (i < n) out[i >> 1] = pack2bf(in[i], in[i + 1]);
}

// h_i = node_emb[nid+1] + lrelu(f @ Wp.T + bp)
// f: nrows x 64 fp32, Wp_bf: 128x64 bf16, out: nrows x 128. nrows % 128 == 0.
__global__ __launch_bounds__(256) void proj_kernel(
    const float* __restrict__ f, const int* __restrict__ nid,
    const uint4* __restrict__ Wp_bf, const float* __restrict__ bp,
    const float* __restrict__ node_emb, float* __restrict__ out)
{
  const int wave = threadIdx.x >> 5;
  const int lane = threadIdx.x & 31;
  const int n16  = lane & 15;
  const int half = lane >> 4;
  const long rowbase = (long)blockIdx.x * 128 + wave * 16;

  // A operand: 16x32 bf16 per K-subtile (K=64 -> 2 subtiles).
  // lanes 0-15 : row M=lane,    K = {k0..k0+7, k0+16..k0+23}
  // lanes 16-31: row M=lane-16, K shifted by +8
  BF16x16 A[2];
  const float* fr = f + (rowbase + n16) * 64;
  #pragma unroll
  for (int t = 0; t < 2; ++t) {
    const float4 a0 = *(const float4*)(fr + t * 32 + half * 8);
    const float4 a1 = *(const float4*)(fr + t * 32 + half * 8 + 4);
    const float4 a2 = *(const float4*)(fr + t * 32 + 16 + half * 8);
    const float4 a3 = *(const float4*)(fr + t * 32 + 16 + half * 8 + 4);
    A[t].u[0] = pack2bf(a0.x, a0.y); A[t].u[1] = pack2bf(a0.z, a0.w);
    A[t].u[2] = pack2bf(a1.x, a1.y); A[t].u[3] = pack2bf(a1.z, a1.w);
    A[t].u[4] = pack2bf(a2.x, a2.y); A[t].u[5] = pack2bf(a2.z, a2.w);
    A[t].u[6] = pack2bf(a3.x, a3.y); A[t].u[7] = pack2bf(a3.z, a3.w);
  }

  #pragma unroll
  for (int j = 0; j < 8; ++j) {            // 8 N-tiles of 16 output cols
    v8f c = {};
    #pragma unroll
    for (int t = 0; t < 2; ++t) {
      // B[k][n] = Wp[n][k]; lane n holds column n, K = t*32 + half*8 + {0..7, 16..23}
      BF16x16 B;
      const int qidx = ((j * 16 + n16) * 64 + t * 32 + half * 8) >> 3;  // 8 bf16/uint4
      B.q[0] = Wp_bf[qidx];
      B.q[1] = Wp_bf[qidx + 2];
      c = __builtin_amdgcn_wmma_f32_16x16x32_bf16(
              false, A[t].v, false, B.v, (short)0, c, false, false);
    }
    const int col = j * 16 + n16;
    const float bias = bp[col];
    #pragma unroll
    for (int v = 0; v < 8; ++v) {
      const long row = rowbase + half * 8 + v;    // C/D layout: half selects M 0-7 / 8-15
      float x = c[v] + bias;
      x = (x > 0.f) ? x : 0.01f * x;              // leaky_relu slope 0.01
      const long g = ((long)nid[row] + 1) * 128 + col;
      out[row * 128 + col] = node_emb[g] + x;
    }
  }
}

// One hop: agg = (sum_{e<5} hs_src[src[5d+e]] - h)/4 ; hn = concat(h,agg)@W.T + b ;
// optional lrelu ; row L2-normalize ; write hs_dst in place. nrows % 128 == 0.
__global__ __launch_bounds__(256) void agg_kernel(
    const float* __restrict__ h_src_buf,   // hs[i]   : Lsrc x 128
    float* __restrict__ h_buf,             // hs[i+1] : nrows x 128, in/out
    const int* __restrict__ src,           // nrows * 5
    const uint4* __restrict__ W_bf,        // 128 x 256 bf16
    const float* __restrict__ b,           // 128
    int apply_lrelu)
{
  __shared__ __align__(16) unsigned int featA[128 * 128]; // 128 rows x 256 bf16 (64 KB)
  __shared__ __align__(16) float hnew[128][132];          // padded fp32 tile (~66 KB)
  const int tid = threadIdx.x;
  const long rowbase = (long)blockIdx.x * 128;

  // ---- phase 1: gather + fixed-fanout segment mean -> bf16 concat tile in LDS
  {
    const int  lrow  = tid >> 1;
    const int  halfc = (tid & 1) * 64;       // this thread's 64-col slab
    const long grow  = rowbase + lrow;
    const float4* hr4 = (const float4*)(h_buf + grow * 128 + halfc);
    const long e0 = grow * 5;
    const float4* sp[5];
    #pragma unroll
    for (int e = 0; e < 5; ++e)
      sp[e] = (const float4*)(h_src_buf + (long)src[e0 + e] * 128 + halfc);
    uint2* dh = (uint2*)(featA + lrow * 128 + (halfc >> 1));       // h   -> cols [0,128)
    uint2* da = (uint2*)(featA + lrow * 128 + 64 + (halfc >> 1));  // agg -> cols [128,256)
    #pragma unroll 4
    for (int c4 = 0; c4 < 16; ++c4) {
      const float4 h4 = hr4[c4];
      float4 s4 = sp[0][c4];
      #pragma unroll
      for (int e = 1; e < 5; ++e) {
        const float4 p = sp[e][c4];
        s4.x += p.x; s4.y += p.y; s4.z += p.z; s4.w += p.w;
      }
      uint2 ph, pa;
      ph.x = pack2bf(h4.x, h4.y);
      ph.y = pack2bf(h4.z, h4.w);
      pa.x = pack2bf((s4.x - h4.x) * 0.25f, (s4.y - h4.y) * 0.25f);  // w==5 -> /(w-1)=4
      pa.y = pack2bf((s4.z - h4.z) * 0.25f, (s4.w - h4.w) * 0.25f);
      dh[c4] = ph;
      da[c4] = pa;
    }
  }
  __syncthreads();

  // ---- phase 2: (16x256) x (256x128) via 64 bf16 WMMAs per wave
  {
    const int wave = tid >> 5, lane = tid & 31;
    const int n16  = lane & 15, half = lane >> 4;
    const uint4* aq = (const uint4*)featA + (wave * 16 + n16) * 32;
    BF16x16 A[8];
    #pragma unroll
    for (int t = 0; t < 8; ++t) {           // ds_load_b128 x2 per K-subtile
      A[t].q[0] = aq[t * 4 + half];         // K = t*32 + half*8 .. +7
      A[t].q[1] = aq[t * 4 + 2 + half];     // K = t*32 + 16 + half*8 .. +7
    }
    #pragma unroll
    for (int j = 0; j < 8; ++j) {
      v8f c = {};
      #pragma unroll
      for (int t = 0; t < 8; ++t) {
        BF16x16 B;
        const int qidx = ((j * 16 + n16) * 256 + t * 32 + half * 8) >> 3;
        B.q[0] = W_bf[qidx];
        B.q[1] = W_bf[qidx + 2];
        c = __builtin_amdgcn_wmma_f32_16x16x32_bf16(
                false, A[t].v, false, B.v, (short)0, c, false, false);
      }
      const int col = j * 16 + n16;
      const float bias = b[col];
      #pragma unroll
      for (int v = 0; v < 8; ++v) {
        float x = c[v] + bias;
        if (apply_lrelu) x = (x > 0.f) ? x : 0.01f * x;
        hnew[wave * 16 + half * 8 + v][col] = x;
      }
    }
  }
  __syncthreads();

  // ---- phase 3: row L2 norm + in-place writeback
  if (tid < 128) {
    const float4* hn = (const float4*)&hnew[tid][0];  // row stride 528 B, 16B aligned
    float ss = 0.f;
    #pragma unroll 8
    for (int c4 = 0; c4 < 32; ++c4) {
      const float4 x = hn[c4];
      ss += x.x * x.x + x.y * x.y + x.z * x.z + x.w * x.w;
    }
    const float r = 1.f / fmaxf(sqrtf(ss), 1e-6f);
    float4* o = (float4*)(h_buf + (rowbase + tid) * 128);
    #pragma unroll 8
    for (int c4 = 0; c4 < 32; ++c4) {
      float4 x = hn[c4];
      x.x *= r; x.y *= r; x.z *= r; x.w *= r;
      o[c4] = x;
    }
  }
}

extern "C" void kernel_launch(void* const* d_in, const int* in_sizes, int n_in,
                              void* d_out, int out_size, void* d_ws, size_t ws_size,
                              hipStream_t stream) {
  // setup_inputs() dict order:
  // 0 node_emb, 1 Wp, 2 bp, 3 W0, 4 b0, 5 W1, 6 b1, 7 W2, 8 b2,
  // 9 f0, 10 nid0, 11 f1, 12 nid1, 13 f2, 14 nid2, 15 f3, 16 nid3,
  // 17 dst0, 18 src0, 19 dst1, 20 src1, 21 dst2, 22 src2
  const float* node_emb = (const float*)d_in[0];
  const float* Wp       = (const float*)d_in[1];
  const float* bp       = (const float*)d_in[2];
  const float* W[3]   = {(const float*)d_in[3], (const float*)d_in[5], (const float*)d_in[7]};
  const float* bb[3]  = {(const float*)d_in[4], (const float*)d_in[6], (const float*)d_in[8]};
  const float* f[4]   = {(const float*)d_in[9],  (const float*)d_in[11],
                         (const float*)d_in[13], (const float*)d_in[15]};
  const int*   nid[4] = {(const int*)d_in[10], (const int*)d_in[12],
                         (const int*)d_in[14], (const int*)d_in[16]};
  const int*   src[3] = {(const int*)d_in[18], (const int*)d_in[20], (const int*)d_in[22]};
  static const int Lsz[4] = {1024000, 204800, 40960, 8192};

  char* ws = (char*)d_ws;
  float* hs[4];
  size_t off = 0;
  for (int i = 0; i < 3; ++i) { hs[i] = (float*)(ws + off); off += (size_t)Lsz[i] * 128 * sizeof(float); }
  hs[3] = (float*)d_out;       // layer-3 initial repr lives in d_out; hop 2 updates it in place

  // bf16 weight staging area at the tail of the workspace
  unsigned int* Wp_bf = (unsigned int*)(ws + off);                 // 128*64  bf16 = 16 KB
  unsigned int* W_bf[3];
  off += (size_t)128 * 64 * 2;
  for (int i = 0; i < 3; ++i) { W_bf[i] = (unsigned int*)(ws + off); off += (size_t)128 * 256 * 2; }

  // pre-convert weights to bf16 (once per launch; cheap, deterministic)
  cvt_bf16_kernel<<<(128 * 64 / 2 + 255) / 256, 256, 0, stream>>>(Wp, Wp_bf, 128 * 64);
  for (int i = 0; i < 3; ++i)
    cvt_bf16_kernel<<<(128 * 256 / 2 + 255) / 256, 256, 0, stream>>>(W[i], W_bf[i], 128 * 256);

  for (int i = 0; i < 4; ++i) {
    const int blocks = Lsz[i] / 128;
    proj_kernel<<<blocks, 256, 0, stream>>>(f[i], nid[i], (const uint4*)Wp_bf, bp,
                                            node_emb, hs[i]);
  }
  for (int i = 0; i < 3; ++i) {
    const int blocks = Lsz[i + 1] / 128;
    agg_kernel<<<blocks, 256, 0, stream>>>(hs[i], hs[i + 1], src[i], (const uint4*)W_bf[i],
                                           bb[i], (i < 2) ? 1 : 0);
  }
  (void)in_sizes; (void)n_in; (void)out_size; (void)ws_size;
}
